// GQARope_53833120088366
// MI455X (gfx1250) — matrix operand order repaired and checked by reading, not compiled, not run on hardware
//
#include <hip/hip_runtime.h>
#include <hip/hip_bf16.h>
#include <cstdint>
#include <cstddef>

// ---------------------------------------------------------------------------
// Problem constants (from the reference): B=2, T=2048, K=1024, H=16, G=2
// ---------------------------------------------------------------------------
static constexpr int Bb  = 2;
static constexpr int Tt  = 2048;
static constexpr int Kk  = 1024;
static constexpr int Hh  = 16;
static constexpr int HGr = 8;      // KV heads = H / G
static constexpr int KH  = 64;     // head dim
static constexpr int QKVN = Kk + 2 * (Kk / 2);   // 2048

#define LOG2E 1.4426950408889634f

typedef __attribute__((ext_vector_type(16))) __bf16 v16bf;
typedef __attribute__((ext_vector_type(8)))  __bf16 v8bf;
typedef __attribute__((ext_vector_type(8)))  float  v8f;

#define WMMA_BF16(a, b, c) \
  __builtin_amdgcn_wmma_f32_16x16x32_bf16(false, (a), false, (b), (short)0, (c), false, false)

__device__ __forceinline__ __bf16 f2bf(float f) {
  unsigned u = __float_as_uint(f);
  u += 0x7FFFu + ((u >> 16) & 1u);              // round-to-nearest-even
  unsigned short s = (unsigned short)(u >> 16);
  return __builtin_bit_cast(__bf16, s);
}

__device__ __forceinline__ v8f zero8() {
  v8f z = {0.f, 0.f, 0.f, 0.f, 0.f, 0.f, 0.f, 0.f};
  return z;
}

// A-matrix fragment (16 x 32 bf16, row-major source).
// lane L: M = L%16; slots 0..15 -> k = 16*(L>=16)+e (contiguous).
__device__ __forceinline__ v16bf load_a16(const __bf16* __restrict__ row, int kb, int hs) {
  const int base = hs * 16;
  v8bf lo = *(const v8bf*)(row + kb + base);
  v8bf hi = *(const v8bf*)(row + kb + base + 8);
  v16bf a;
#pragma unroll
  for (int i = 0; i < 8; ++i) { a[i] = lo[i]; a[i + 8] = hi[i]; }
  return a;
}

// ---------------------------------------------------------------------------
// fp32 -> bf16 elementwise convert
// ---------------------------------------------------------------------------
__global__ void conv_f32_bf16(const float* __restrict__ src, __bf16* __restrict__ dst, int n) {
  int i = blockIdx.x * 256 + threadIdx.x;
  if (i < n) dst[i] = f2bf(src[i]);
}

// src (R x C) fp32 row-major  ->  dst (C x R) bf16 row-major (i.e. B transposed)
__global__ void transpose_conv(const float* __restrict__ src, __bf16* __restrict__ dst,
                               int R, int C) {
  int i = blockIdx.x * 256 + threadIdx.x;
  if (i < R * C) {
    int c = i / R, r = i - c * R;
    dst[i] = f2bf(src[(size_t)r * C + c]);
  }
}

// ---------------------------------------------------------------------------
// Wave-level bf16 WMMA GEMM:  C(MxN,f32) = A(MxKd,bf16) * Bt(NxKd,bf16)^T
// grid = (M/32, N/64), block = 32 (one wave per 32x64 tile).
// Ping-pong double buffering (unroll-by-2, no fragment copies): while the 8
// WMMAs of one k-chunk execute, the loads for the next chunk are in flight.
// Requires Kd to be a multiple of 64 (holds: Kd == 1024).
// ---------------------------------------------------------------------------
struct Frags { v16bf a0, a1, b0, b1, b2, b3; };

__device__ __forceinline__ void load_frags(Frags& f,
                                           const __bf16* __restrict__ arow0,
                                           const __bf16* __restrict__ arow1,
                                           const __bf16* __restrict__ brow,
                                           int kb, int hs, int Kd) {
  f.a0 = load_a16(arow0, kb, hs);
  f.a1 = load_a16(arow1, kb, hs);
  f.b0 = *(const v16bf*)(brow + kb);
  f.b1 = *(const v16bf*)(brow + kb + (size_t)16 * Kd);
  f.b2 = *(const v16bf*)(brow + kb + (size_t)32 * Kd);
  f.b3 = *(const v16bf*)(brow + kb + (size_t)48 * Kd);
}

__device__ __forceinline__ void mma8(v8f acc[8], const Frags& f) {
  acc[0] = WMMA_BF16(f.a0, f.b0, acc[0]);
  acc[1] = WMMA_BF16(f.a0, f.b1, acc[1]);
  acc[2] = WMMA_BF16(f.a0, f.b2, acc[2]);
  acc[3] = WMMA_BF16(f.a0, f.b3, acc[3]);
  acc[4] = WMMA_BF16(f.a1, f.b0, acc[4]);
  acc[5] = WMMA_BF16(f.a1, f.b1, acc[5]);
  acc[6] = WMMA_BF16(f.a1, f.b2, acc[6]);
  acc[7] = WMMA_BF16(f.a1, f.b3, acc[7]);
}

__global__ void gemm_bf16_f32(const __bf16* __restrict__ A, const __bf16* __restrict__ Bt,
                              float* __restrict__ C, int N, int Kd) {
  const int lane = threadIdx.x;
  const int lm = lane & 15, hs = lane >> 4;
  const int m0 = blockIdx.x * 32;
  const int n0 = blockIdx.y * 64;

  const __bf16* arow0 = A + (size_t)(m0 + lm) * Kd;
  const __bf16* arow1 = A + (size_t)(m0 + 16 + lm) * Kd;
  const __bf16* brow  = Bt + (size_t)(n0 + lm) * Kd + hs * 16;

  v8f acc[8];
#pragma unroll
  for (int i = 0; i < 8; ++i) acc[i] = zero8();

  Frags fA, fB;
  load_frags(fA, arow0, arow1, brow, 0, hs, Kd);

  int kb = 0;
  for (; kb + 64 < Kd; kb += 64) {
    __builtin_prefetch(arow0 + kb + 96, 0, 3);   // global_prefetch_b8
    __builtin_prefetch(brow  + kb + 96, 0, 3);
    load_frags(fB, arow0, arow1, brow, kb + 32, hs, Kd);
    mma8(acc, fA);
    load_frags(fA, arow0, arow1, brow, kb + 64, hs, Kd);
    mma8(acc, fB);
  }
  // tail: kb == Kd - 64
  load_frags(fB, arow0, arow1, brow, kb + 32, hs, Kd);
  mma8(acc, fA);
  mma8(acc, fB);

#pragma unroll
  for (int s = 0; s < 2; ++s)
#pragma unroll
    for (int r = 0; r < 8; ++r) {
      size_t o = (size_t)(m0 + s * 16 + r + hs * 8) * N + n0 + lm;   // C layout
      C[o]      = acc[s * 4 + 0][r];
      C[o + 16] = acc[s * 4 + 1][r];
      C[o + 32] = acc[s * 4 + 2][r];
      C[o + 48] = acc[s * 4 + 3][r];
    }
}

// ---------------------------------------------------------------------------
// RoPE + head split.  One thread per (b, head, t, d); head 0..15 = Q, 16..23 = K
// Qr: (B,H,T,64) bf16 ;  Kr: (B,8,T,64) bf16
// ---------------------------------------------------------------------------
__global__ void rope_kernel(const float* __restrict__ qkv,
                            const float* __restrict__ cosb, const float* __restrict__ sinb,
                            __bf16* __restrict__ Qr, __bf16* __restrict__ Kr) {
  int i = blockIdx.x * 256 + threadIdx.x;
  const int total = Bb * (Hh + HGr) * Tt * KH;
  if (i >= total) return;
  int d = i & 63;
  int r = i >> 6;
  int t = r % Tt;  r /= Tt;
  int head = r % (Hh + HGr);
  int b = r / (Hh + HGr);

  int col = (head < Hh) ? head * KH + d : Kk + (head - Hh) * KH + d;
  const float* row = qkv + ((size_t)b * Tt + t) * QKVN;
  float v  = row[col];
  int dp   = (d < 32) ? d + 32 : d - 32;
  float vp = row[col - d + dp];
  float rot = (d < 32) ? -vp : vp;
  float out = v * cosb[t * KH + d] + rot * sinb[t * KH + d];

  if (head < Hh)
    Qr[((size_t)(b * Hh + head) * Tt + t) * KH + d] = f2bf(out);
  else
    Kr[((size_t)(b * HGr + head - Hh) * Tt + t) * KH + d] = f2bf(out);
}

// V stored d-major: Vt (B, 8, 64, T) bf16 — makes the P*V B-fragment contiguous
__global__ void vtrans_kernel(const float* __restrict__ qkv, __bf16* __restrict__ Vt) {
  int i = blockIdx.x * 256 + threadIdx.x;
  const int total = Bb * HGr * KH * Tt;
  if (i >= total) return;
  int t = i % Tt;
  int r = i / Tt;
  int d = r & 63; r >>= 6;
  int kvh = r % HGr;
  int b = r / HGr;
  Vt[(((size_t)b * HGr + kvh) * KH + d) * Tt + t] =
      f2bf(qkv[((size_t)b * Tt + t) * QKVN + Kk + (Kk / 2) + kvh * KH + d]);
}

// ---------------------------------------------------------------------------
// Flash attention: one wave per (b,h, 32-query tile). Online softmax over
// 32-key blocks, causal. K/V fragments are shared by both 16-row query
// sub-tiles (16 WMMAs per key block). Each sub-tile is processed end-to-end
// to keep the probability registers short-lived. Oo: (B,T,1024) bf16.
// ---------------------------------------------------------------------------
__global__ void attn_kernel(const __bf16* __restrict__ Qr, const __bf16* __restrict__ Kr,
                            const __bf16* __restrict__ Vt, __bf16* __restrict__ Oo) {
  const int lane = threadIdx.x;
  const int lm = lane & 15, hs = lane >> 4;
  const int bh  = blockIdx.y;            // 0..31
  const int b   = bh >> 4, h = bh & 15;
  const int bhk = b * HGr + (h >> 1);    // GQA: 2 query heads per KV head
  const int q0  = blockIdx.x * 32;       // multiple of 32

  __shared__ __bf16 Plds[32 * 32];       // C-layout -> A-layout exchange

  v16bf aq[2][2];
#pragma unroll
  for (int s = 0; s < 2; ++s) {
    const __bf16* qrow = Qr + ((size_t)bh * Tt + q0 + s * 16 + lm) * KH;
    aq[s][0] = load_a16(qrow, 0, hs);    // d = 0..31
    aq[s][1] = load_a16(qrow, 32, hs);   // d = 32..63
  }

  float m[16], l[16];
#pragma unroll
  for (int r = 0; r < 16; ++r) { m[r] = -3.0e38f; l[r] = 0.f; }
  v8f acc[2][4];
#pragma unroll
  for (int s = 0; s < 2; ++s)
#pragma unroll
    for (int dt = 0; dt < 4; ++dt) acc[s][dt] = zero8();

  for (int kb = 0; kb <= q0 + 31; kb += 32) {
    // ---- K fragments for keys kb..kb+31 (shared by both query sub-tiles) ----
    const __bf16* k0row = Kr + ((size_t)bhk * Tt + kb + lm) * KH + hs * 16;
    v16bf bk00 = *(const v16bf*)(k0row);
    v16bf bk01 = *(const v16bf*)(k0row + 32);
    v16bf bk10 = *(const v16bf*)(k0row + (size_t)16 * KH);
    v16bf bk11 = *(const v16bf*)(k0row + (size_t)16 * KH + 32);

    v16bf ap[2];
#pragma unroll
    for (int s = 0; s < 2; ++s) {
      // ---- S = Q K^T for this sub-tile : 4 WMMAs ----
      v8f sc0 = zero8(), sc1 = zero8();
      sc0 = WMMA_BF16(aq[s][0], bk00, sc0);
      sc0 = WMMA_BF16(aq[s][1], bk01, sc0);
      sc1 = WMMA_BF16(aq[s][0], bk10, sc1);
      sc1 = WMMA_BF16(aq[s][1], bk11, sc1);

      // ---- scale + causal mask (C layout: lane=key col, reg=row) ----
      float p0[8], p1[8], rm[8];
#pragma unroll
      for (int r = 0; r < 8; ++r) {
        int row = q0 + s * 16 + r + hs * 8;
        float v0 = (kb + lm      <= row) ? sc0[r] * 0.125f : -3.0e38f;
        float v1 = (kb + 16 + lm <= row) ? sc1[r] * 0.125f : -3.0e38f;
        p0[r] = v0; p1[r] = v1;
        rm[r] = fmaxf(v0, v1);
      }
#pragma unroll
      for (int r = 0; r < 8; ++r)
#pragma unroll
        for (int msk = 1; msk < 16; msk <<= 1)
          rm[r] = fmaxf(rm[r], __shfl_xor(rm[r], msk, 32));

      // ---- online softmax update ----
      float alpha[8];
#pragma unroll
      for (int r = 0; r < 8; ++r) {
        int i = s * 8 + r;
        float mn = fmaxf(m[i], rm[r]);
        alpha[r] = __builtin_exp2f((m[i] - mn) * LOG2E);
        m[i] = mn;
        p0[r] = __builtin_exp2f((p0[r] - mn) * LOG2E);
        p1[r] = __builtin_exp2f((p1[r] - mn) * LOG2E);
      }
#pragma unroll
      for (int r = 0; r < 8; ++r) {
        float rs = p0[r] + p1[r];
#pragma unroll
        for (int msk = 1; msk < 16; msk <<= 1)
          rs += __shfl_xor(rs, msk, 32);
        l[s * 8 + r] = l[s * 8 + r] * alpha[r] + rs;
      }
#pragma unroll
      for (int dt = 0; dt < 4; ++dt)
#pragma unroll
        for (int r = 0; r < 8; ++r) acc[s][dt][r] *= alpha[r];

      // ---- P: C layout -> A layout through LDS (rows s*16 .. s*16+15) ----
      __syncthreads();
#pragma unroll
      for (int r = 0; r < 8; ++r) {
        Plds[(s * 16 + r + hs * 8) * 32 + lm]      = f2bf(p0[r]);
        Plds[(s * 16 + r + hs * 8) * 32 + 16 + lm] = f2bf(p1[r]);
      }
      __syncthreads();
#pragma unroll
      for (int e = 0; e < 8; ++e) {
        ap[s][e]     = Plds[(s * 16 + lm) * 32 + hs * 16 + e];        // kk = 16*hs + e
        ap[s][e + 8] = Plds[(s * 16 + lm) * 32 + hs * 16 + 8 + e];   // kk = 16*hs + 8 + e
      }
    }

    // ---- O += P * V : 8 WMMAs (V fragments shared by both sub-tiles) ----
    const __bf16* vbase = Vt + ((size_t)bhk * KH + lm) * Tt + kb + hs * 16;
#pragma unroll
    for (int dt = 0; dt < 4; ++dt) {
      v16bf bv = *(const v16bf*)(vbase + (size_t)(dt * 16) * Tt);
      acc[0][dt] = WMMA_BF16(ap[0], bv, acc[0][dt]);
      acc[1][dt] = WMMA_BF16(ap[1], bv, acc[1][dt]);
    }
  }

  // ---- normalize + store (B,T,H*64) bf16 ----
#pragma unroll
  for (int s = 0; s < 2; ++s)
#pragma unroll
    for (int r = 0; r < 8; ++r) {
      float inv = 1.0f / l[s * 8 + r];
      int row = q0 + s * 16 + r + hs * 8;
      __bf16* op = Oo + ((size_t)b * Tt + row) * Kk + h * KH + lm;
#pragma unroll
      for (int dt = 0; dt < 4; ++dt) op[dt * 16] = f2bf(acc[s][dt][r] * inv);
    }
}

// ---------------------------------------------------------------------------
// Host-side launch
// ---------------------------------------------------------------------------
extern "C" void kernel_launch(void* const* d_in, const int* in_sizes, int n_in,
                              void* d_out, int out_size, void* d_ws, size_t ws_size,
                              hipStream_t stream) {
  const float* x     = (const float*)d_in[0];
  const float* cosb  = (const float*)d_in[1];
  const float* sinb  = (const float*)d_in[2];
  const float* w_qkv = (const float*)d_in[3];
  const float* w_out = (const float*)d_in[4];
  float* out = (float*)d_out;

  char* ws = (char*)d_ws;
  __bf16* x_bf   = (__bf16*)(ws);                         //  8 MiB: (4096,1024) bf16
  __bf16* wqkv_t = (__bf16*)(ws + ( 8ull << 20));         //  4 MiB: (2048,1024) bf16 (W^T)
  __bf16* wout_t = (__bf16*)(ws + (12ull << 20));         //  2 MiB: (1024,1024) bf16 (W^T)
  float*  qkv    = (float*) (ws + (14ull << 20));         // 32 MiB: (4096,2048) f32
  __bf16* attnO  = (__bf16*)(ws + (14ull << 20));         //  8 MiB, reuses qkv region
  __bf16* Qr     = (__bf16*)(ws + (46ull << 20));         //  8 MiB: (2,16,2048,64)
  __bf16* Kr     = (__bf16*)(ws + (54ull << 20));         //  4 MiB: (2, 8,2048,64)
  __bf16* Vt     = (__bf16*)(ws + (58ull << 20));         //  4 MiB: (2, 8,  64,2048)

  { int n = Bb * Tt * Kk;
    conv_f32_bf16<<<(n + 255) / 256, 256, 0, stream>>>(x, x_bf, n); }
  { int n = Kk * QKVN;
    transpose_conv<<<(n + 255) / 256, 256, 0, stream>>>(w_qkv, wqkv_t, Kk, QKVN); }
  { int n = Kk * Kk;
    transpose_conv<<<(n + 255) / 256, 256, 0, stream>>>(w_out, wout_t, Kk, Kk); }

  gemm_bf16_f32<<<dim3((Bb * Tt) / 32, QKVN / 64), 32, 0, stream>>>(x_bf, wqkv_t, qkv, QKVN, Kk);

  { int n = Bb * (Hh + HGr) * Tt * KH;
    rope_kernel<<<(n + 255) / 256, 256, 0, stream>>>(qkv, cosb, sinb, Qr, Kr); }
  { int n = Bb * HGr * KH * Tt;
    vtrans_kernel<<<(n + 255) / 256, 256, 0, stream>>>(qkv, Vt); }

  attn_kernel<<<dim3(Tt / 32, Bb * Hh), 32, 0, stream>>>(Qr, Kr, Vt, attnO);

  gemm_bf16_f32<<<dim3((Bb * Tt) / 32, Kk / 64), 32, 0, stream>>>(attnO, wout_t, out, Kk, Kk);
}
